// GAT_9371618640567
// MI455X (gfx1250) — matrix operand order repaired and checked by reading, not compiled, not run on hardware
//
#include <hip/hip_runtime.h>
#include <hip/hip_bf16.h>

#define N_NODES 50000
#define E_EDGES 800000
#define ETOT    (E_EDGES + N_NODES)   // reference appends one self-loop per node
#define HC      128
#define KDIM    128                   // K is 128 for all three GEMMs
#define HEADS   4
#define CHID    32
#define OUT_DIM 40
#define OUT_PAD 48
#define EPS_BN  1e-5f

typedef __attribute__((ext_vector_type(16))) __bf16 v16bf;
typedef __attribute__((ext_vector_type(8)))  __bf16 v8bf;
typedef __attribute__((ext_vector_type(8)))  float  v8f;

// ---------- helpers ----------
__device__ __forceinline__ float lrelu02(float v) { return v > 0.f ? v : 0.2f * v; }

// Monotonic float<->uint mapping so unsigned atomicMax implements float max.
__device__ __forceinline__ unsigned fmapOrd(float f) {
    unsigned u = __float_as_uint(f);
    return (u & 0x80000000u) ? ~u : (u | 0x80000000u);
}
__device__ __forceinline__ float funmapOrd(unsigned m) {
    unsigned u = (m & 0x80000000u) ? (m & 0x7fffffffu) : ~m;
    return __uint_as_float(u);
}
#define EMAX_NEG_INF 0x007fffffu   // fmapOrd(-inf)

__device__ __forceinline__ void edge_sd(const int* __restrict__ ei, int e, int& s, int& d) {
    if (e < E_EDGES) { s = ei[e]; d = ei[E_EDGES + e]; }
    else             { s = d = e - E_EDGES; }       // self-loop
}

// ---------- 0a) f32 -> bf16 convert ----------
__global__ __launch_bounds__(256) void cvt_bf16_k(
    const float* __restrict__ in, __bf16* __restrict__ out, int n)
{
    const int i = blockIdx.x * blockDim.x + threadIdx.x;
    if (i < n) out[i] = (__bf16)in[i];
}

// ---------- 0b) pack W[K x Ncol] (f32) into bf16 B-fragment order, zero padded ----------
// Fragment element e (0..15) of (tile, kb, lane):  k = kb*32 + 16*(e>>3) + 8*(lane>>4) + (e&7),
//                                                  n = tile*16 + (lane&15)
__global__ __launch_bounds__(256) void pack_w_k(
    const float* __restrict__ W, __bf16* __restrict__ Wfrag, int Ncol, int tilesN)
{
    const int i = blockIdx.x * blockDim.x + threadIdx.x;      // bf16 element index
    const int total = tilesN * (KDIM / 32) * 32 * 16;
    if (i >= total) return;
    const int e    = i & 15;
    const int lane = (i >> 4) & 31;
    const int kb   = (i >> 9) % (KDIM / 32);
    const int tile = (i >> 9) / (KDIM / 32);
    const int k = kb * 32 + ((e >> 3) * 16) + ((lane >> 4) * 8) + (e & 7);
    const int n = tile * 16 + (lane & 15);
    const float v = (n < Ncol) ? W[(size_t)k * Ncol + n] : 0.f;
    Wfrag[i] = (__bf16)v;
}

// ---------- 1) GEMM via bf16 WMMA, LDS-staged: H[Mx Npad] = Xbf[M x 128] * W ----------
// One block per 16-row tile (gridDim.x = M/16); wave w computes column tile w.
__global__ __launch_bounds__(256) void gat_gemm_wmma(
    const __bf16* __restrict__ Xbf, const __bf16* __restrict__ Wfrag,
    float* __restrict__ H, int tilesN, int Npad)
{
    __shared__ __align__(16) __bf16 shA[16 * KDIM];                 // 4 KB
    __shared__ __align__(16) __bf16 shW[8 * (KDIM / 32) * 32 * 16]; // up to 32 KB

    const int tid = threadIdx.x;
    const int m0  = blockIdx.x * 16;

    // cooperative stage: A tile (16 rows x 128 bf16), 16B per chunk, 256 chunks
    {
        const int cpr = KDIM / 8;                       // 16B chunks per row
        for (int i = tid; i < 16 * cpr; i += 256) {
            const int row = i / cpr, c = i % cpr;
            reinterpret_cast<uint4*>(shA)[i] =
                *reinterpret_cast<const uint4*>(Xbf + (size_t)(m0 + row) * KDIM + c * 8);
        }
        // packed weight fragments for all column tiles
        const int wchunks = tilesN * (KDIM / 32) * 32 * 16 / 8;   // 16B chunks
        for (int i = tid; i < wchunks; i += 256) {
            reinterpret_cast<uint4*>(shW)[i] = reinterpret_cast<const uint4*>(Wfrag)[i];
        }
    }
    __syncthreads();

    const int wv   = tid >> 5;
    const int lane = tid & 31;
    if (wv < tilesN) {                                  // wave-uniform: EXEC all-ones inside
        const int half = lane >> 4;
        const int l15  = lane & 15;
        const __bf16* arow = shA + l15 * KDIM;

        v8f acc = {};
#pragma unroll
        for (int kb = 0; kb < KDIM / 32; ++kb) {
            const int k0 = kb * 32;
            // A 16x32 bf16 fragment: two contiguous 8-elem runs per lane
            const v8bf alo = *reinterpret_cast<const v8bf*>(arow + k0 + 8 * half);
            const v8bf ahi = *reinterpret_cast<const v8bf*>(arow + k0 + 16 + 8 * half);
            const v16bf a = __builtin_shufflevector(alo, ahi,
                0, 1, 2, 3, 4, 5, 6, 7, 8, 9, 10, 11, 12, 13, 14, 15);
            // B 32x16 bf16 fragment: pre-packed contiguous 16 elems per lane
            const __bf16* wp = shW + (size_t)(((wv * (KDIM / 32) + kb) * 32 + lane) * 16);
            const v8bf blo = *reinterpret_cast<const v8bf*>(wp);
            const v8bf bhi = *reinterpret_cast<const v8bf*>(wp + 8);
            const v16bf b = __builtin_shufflevector(blo, bhi,
                0, 1, 2, 3, 4, 5, 6, 7, 8, 9, 10, 11, 12, 13, 14, 15);
            acc = __builtin_amdgcn_wmma_f32_16x16x32_bf16(
                      false, a, false, b, (short)0, acc, false, false);
        }
#pragma unroll
        for (int r = 0; r < 8; ++r) {                   // C/D: VGPR r -> M=r (+8 hi half)
            H[(size_t)(m0 + r + half * 8) * Npad + wv * 16 + l15] = acc[r];
        }
    }
}

// ---------- 2) per-(node,head) attention logits ----------
__global__ __launch_bounds__(256) void attn_scores_k(
    const float* __restrict__ Hl, const float* __restrict__ asv, const float* __restrict__ adv,
    float* __restrict__ es, float* __restrict__ ed, int nnodes, int heads, int ch, int ldh)
{
    const int i = blockIdx.x * blockDim.x + threadIdx.x;
    if (i >= nnodes * heads) return;
    const int n = i / heads, h = i - n * heads;
    const float* row = Hl + (size_t)n * ldh + h * ch;
    float s = 0.f, d = 0.f;
    for (int c = 0; c < ch; ++c) {
        const float v = row[c];
        s += v * asv[h * ch + c];
        d += v * adv[h * ch + c];
    }
    es[i] = s; ed[i] = d;
}

// ---------- 3) init aggregate (=bias), segment-max, denominators ----------
__global__ __launch_bounds__(256) void init_layer_k(
    float* __restrict__ agg, unsigned* __restrict__ emax, float* __restrict__ denom,
    const float* __restrict__ bias, int nnodes, int heads, int ncol, int ldagg)
{
    const int i = blockIdx.x * blockDim.x + threadIdx.x;
    if (i < nnodes * ldagg) {
        const int c = i % ldagg;
        agg[i] = (c < ncol) ? bias[c] : 0.f;
    }
    if (i < nnodes * heads) {
        emax[i]  = EMAX_NEG_INF;
        denom[i] = 0.f;
    }
}

// ---------- 4) segment max over dst ----------
__global__ __launch_bounds__(256) void edge_max_k(
    const int* __restrict__ ei, const float* __restrict__ es, const float* __restrict__ ed,
    unsigned* __restrict__ emax, int heads)
{
    const int i = blockIdx.x * blockDim.x + threadIdx.x;
    if (i >= ETOT * heads) return;
    const int e = i / heads, h = i - e * heads;
    int s, d; edge_sd(ei, e, s, d);
    const float v = lrelu02(es[s * heads + h] + ed[d * heads + h]);
    atomicMax(&emax[d * heads + h], fmapOrd(v));
}

// ---------- 5) exp + segment sum ----------
__global__ __launch_bounds__(256) void edge_exp_k(
    const int* __restrict__ ei, const float* __restrict__ es, const float* __restrict__ ed,
    const unsigned* __restrict__ emax, float* __restrict__ denom, float* __restrict__ exb, int heads)
{
    const int i = blockIdx.x * blockDim.x + threadIdx.x;
    if (i >= ETOT * heads) return;
    const int e = i / heads, h = i - e * heads;
    int s, d; edge_sd(ei, e, s, d);
    const float v = lrelu02(es[s * heads + h] + ed[d * heads + h]);
    const float ex = expf(v - funmapOrd(emax[d * heads + h]));
    exb[i] = ex;
    atomicAdd(&denom[d * heads + h], ex);
}

// ---------- 6) alpha-weighted scatter aggregation: one wave per edge ----------
__global__ __launch_bounds__(256) void edge_aggr_k(
    const int* __restrict__ ei, const float* __restrict__ Hl,
    const float* __restrict__ exb, const float* __restrict__ denom,
    float* __restrict__ agg, int heads, int ch, int ldh, int ldagg)
{
    const int wave = blockIdx.x * (blockDim.x >> 5) + (threadIdx.x >> 5);
    const int lane = threadIdx.x & 31;
    if (wave >= ETOT) return;
    int s, d; edge_sd(ei, wave, s, d);
    for (int h = 0; h < heads; ++h) {
        const float alpha = exb[wave * heads + h] / denom[d * heads + h];
        for (int c = lane; c < ch; c += 32) {
            const float val = Hl[(size_t)s * ldh + h * ch + c] * alpha;
            atomicAdd(&agg[(size_t)d * ldagg + h * ch + c], val);
        }
    }
}

// ---------- 7) batch-norm stats: one block per channel ----------
__global__ __launch_bounds__(256) void bn_stats_k(
    const float* __restrict__ Hin, float* __restrict__ mu, float* __restrict__ var,
    int nrows, int ld)
{
    __shared__ float sh_s[256];
    __shared__ float sh_q[256];
    const int c = blockIdx.x;
    float s = 0.f, q = 0.f;
    for (int r = threadIdx.x; r < nrows; r += 256) {
        const float v = Hin[(size_t)r * ld + c];
        s += v; q += v * v;
    }
    sh_s[threadIdx.x] = s; sh_q[threadIdx.x] = q;
    __syncthreads();
    for (int o = 128; o > 0; o >>= 1) {
        if ((int)threadIdx.x < o) {
            sh_s[threadIdx.x] += sh_s[threadIdx.x + o];
            sh_q[threadIdx.x] += sh_q[threadIdx.x + o];
        }
        __syncthreads();
    }
    if (threadIdx.x == 0) {
        const float m = sh_s[0] / (float)nrows;
        mu[c]  = m;
        var[c] = sh_q[0] / (float)nrows - m * m;
    }
}

// ---------- 8) BN + ReLU elementwise, emits bf16 (next layer's GEMM input) ----------
__global__ __launch_bounds__(256) void bn_relu_k(
    const float* __restrict__ Hin, const float* __restrict__ mu, const float* __restrict__ var,
    const float* __restrict__ gamma, const float* __restrict__ beta,
    __bf16* __restrict__ Hout, int nrows, int ld)
{
    const int i = blockIdx.x * blockDim.x + threadIdx.x;
    if (i >= nrows * ld) return;
    const int c = i % ld;
    float v = (Hin[i] - mu[c]) * rsqrtf(var[c] + EPS_BN) * gamma[c] + beta[c];
    Hout[i] = (__bf16)(v > 0.f ? v : 0.f);
}

// ---------- host-side layer driver ----------
static void gat_layer(const __bf16* inBf, const int* ei,
                      const float* W, __bf16* Wfrag,
                      const float* asv, const float* adv, const float* bias,
                      int heads, int ch, int ncol, int npad,
                      float* Hlin, float* agg, int ldagg,
                      float* es, float* ed, unsigned* emax, float* denom, float* exb,
                      hipStream_t stream)
{
    const int tilesN = npad / 16;
    const int wtotal = tilesN * (KDIM / 32) * 32 * 16;
    pack_w_k<<<dim3((wtotal + 255) / 256), dim3(256), 0, stream>>>(W, Wfrag, ncol, tilesN);

    gat_gemm_wmma<<<dim3(N_NODES / 16), dim3(256), 0, stream>>>(
        inBf, Wfrag, Hlin, tilesN, npad);

    const int nh = N_NODES * heads;
    attn_scores_k<<<dim3((nh + 255) / 256), dim3(256), 0, stream>>>(
        Hlin, asv, adv, es, ed, N_NODES, heads, ch, npad);

    int initN = N_NODES * ldagg; if (nh > initN) initN = nh;
    init_layer_k<<<dim3((initN + 255) / 256), dim3(256), 0, stream>>>(
        agg, emax, denom, bias, N_NODES, heads, ncol, ldagg);

    const int eh = ETOT * heads;
    edge_max_k<<<dim3((eh + 255) / 256), dim3(256), 0, stream>>>(ei, es, ed, emax, heads);
    edge_exp_k<<<dim3((eh + 255) / 256), dim3(256), 0, stream>>>(ei, es, ed, emax, denom, exb, heads);
    edge_aggr_k<<<dim3((ETOT + 7) / 8), dim3(256), 0, stream>>>(
        ei, Hlin, exb, denom, agg, heads, ch, npad, ldagg);
}

extern "C" void kernel_launch(void* const* d_in, const int* in_sizes, int n_in,
                              void* d_out, int out_size, void* d_ws, size_t ws_size,
                              hipStream_t stream)
{
    (void)in_sizes; (void)n_in; (void)out_size; (void)ws_size;

    const float* x   = (const float*)d_in[0];
    const int*   ei  = (const int*)d_in[1];
    const float* W0  = (const float*)d_in[2];
    const float* as0 = (const float*)d_in[3];
    const float* ad0 = (const float*)d_in[4];
    const float* b0  = (const float*)d_in[5];
    const float* g0  = (const float*)d_in[6];
    const float* be0 = (const float*)d_in[7];
    const float* W1  = (const float*)d_in[8];
    const float* as1 = (const float*)d_in[9];
    const float* ad1 = (const float*)d_in[10];
    const float* b1  = (const float*)d_in[11];
    const float* g1  = (const float*)d_in[12];
    const float* be1 = (const float*)d_in[13];
    const float* W2  = (const float*)d_in[14];
    const float* as2 = (const float*)d_in[15];
    const float* ad2 = (const float*)d_in[16];
    const float* b2  = (const float*)d_in[17];
    float* out = (float*)d_out;

    // workspace carving (all offsets 256B aligned; ~92 MB total)
    char* ws = (char*)d_ws;
    size_t off = 0;
    auto carve = [&](size_t bytes) -> char* {
        char* p = ws + off;
        off = (off + bytes + 255) & ~(size_t)255;
        return p;
    };
    float*    A     = (float*)   carve((size_t)N_NODES * HC * 4);   // GEMM output h_lin (f32)
    float*    B     = (float*)   carve((size_t)N_NODES * HC * 4);   // aggregate buffer (f32)
    __bf16*   Xbf   = (__bf16*)  carve((size_t)N_NODES * HC * 2);   // layer-0 input, bf16
    __bf16*   Cbf   = (__bf16*)  carve((size_t)N_NODES * HC * 2);   // BN+ReLU output, bf16
    __bf16*   Wfrag = (__bf16*)  carve((size_t)8 * (KDIM / 32) * 32 * 16 * 2); // 32 KB
    float*    es    = (float*)   carve((size_t)N_NODES * HEADS * 4);
    float*    ed    = (float*)   carve((size_t)N_NODES * HEADS * 4);
    unsigned* emax  = (unsigned*)carve((size_t)N_NODES * HEADS * 4);
    float*    denom = (float*)   carve((size_t)N_NODES * HEADS * 4);
    float*    exb   = (float*)   carve((size_t)ETOT * HEADS * 4);
    float*    mu    = (float*)   carve(HC * 4);
    float*    var   = (float*)   carve(HC * 4);

    // x -> bf16 once
    cvt_bf16_k<<<dim3((N_NODES * HC + 255) / 256), dim3(256), 0, stream>>>(
        x, Xbf, N_NODES * HC);

    // ----- layer 0 -----
    gat_layer(Xbf, ei, W0, Wfrag, as0, ad0, b0, HEADS, CHID, HC, HC,
              A, B, HC, es, ed, emax, denom, exb, stream);
    bn_stats_k<<<dim3(HC), dim3(256), 0, stream>>>(B, mu, var, N_NODES, HC);
    bn_relu_k<<<dim3((N_NODES * HC + 255) / 256), dim3(256), 0, stream>>>(
        B, mu, var, g0, be0, Cbf, N_NODES, HC);

    // ----- layer 1 -----
    gat_layer(Cbf, ei, W1, Wfrag, as1, ad1, b1, HEADS, CHID, HC, HC,
              A, B, HC, es, ed, emax, denom, exb, stream);
    bn_stats_k<<<dim3(HC), dim3(256), 0, stream>>>(B, mu, var, N_NODES, HC);
    bn_relu_k<<<dim3((N_NODES * HC + 255) / 256), dim3(256), 0, stream>>>(
        B, mu, var, g1, be1, Cbf, N_NODES, HC);

    // ----- layer 2 (heads=1, 40 classes, aggregate straight into d_out) -----
    gat_layer(Cbf, ei, W2, Wfrag, as2, ad2, b2, 1, OUT_DIM, OUT_DIM, OUT_PAD,
              A, out, OUT_DIM, es, ed, emax, denom, exb, stream);
}
